// MC_Model_53188874994091
// MI455X (gfx1250) — matrix operand order repaired and checked by reading, not compiled, not run on hardware
//
#include <hip/hip_runtime.h>
#include <hip/hip_bf16.h>
#include <math.h>

typedef _Float16 v16h __attribute__((ext_vector_type(16)));
typedef _Float16 v8h  __attribute__((ext_vector_type(8)));
typedef float    v8f  __attribute__((ext_vector_type(8)));

#define NEAR_F 2.0f
#define FAR_F  6.0f
#define SC     64
#define SF     256
#define NRAYS  4096
#define WPB    4          // waves per block (128 threads)
#define DIN    66
#define HID    64

__device__ __forceinline__ float softplusf_(float x) {
    return (x > 0.f) ? (x + log1pf(__expf(-x))) : log1pf(__expf(x));
}
__device__ __forceinline__ float sigmoidf_(float x) {
    return 1.f / (1.f + __expf(-x));
}

// ---- WMMA fragment holders -------------------------------------------------
struct MLPW {
    v16h B1[3][4];   // layer1 weights: [k_step 0..2][n_tile 0..3], K padded 66->96
    v16h B2[2];      // layer2 weights: [k_step 0..1], N padded 4->16
    float b1v[4];    // bias1 for this lane's N (per n_tile)
    float b2v;       // bias2 for this lane's N (0 if n>=4)
};

__device__ __forceinline__ void load_weights(MLPW& W, const float* __restrict__ w1,
                                             const float* __restrict__ b1,
                                             const float* __restrict__ w2,
                                             const float* __restrict__ b2, int lane) {
    const int nl  = lane & 15;
    const int hi  = lane >> 4;
    #pragma unroll
    for (int ks = 0; ks < 3; ++ks) {
        #pragma unroll
        for (int nt = 0; nt < 4; ++nt) {
            const int n  = nt * 16 + nl;
            const int kb = ks * 32 + hi * 8;        // B layout: lanes16-31 hold K+8 (table order)
            v16h b;
            #pragma unroll
            for (int h = 0; h < 16; ++h) {
                const int k = kb + (h < 8 ? h : h + 8);
                b[h] = (_Float16)((k < DIN) ? w1[k * HID + n] : 0.f);
            }
            W.B1[ks][nt] = b;
        }
    }
    #pragma unroll
    for (int ks = 0; ks < 2; ++ks) {
        const int kb = ks * 32 + hi * 8;
        v16h b;
        #pragma unroll
        for (int h = 0; h < 16; ++h) {
            const int k = kb + (h < 8 ? h : h + 8); // k < 64 always
            b[h] = (_Float16)((nl < 4) ? w2[k * 4 + nl] : 0.f);
        }
        W.B2[ks] = b;
    }
    #pragma unroll
    for (int nt = 0; nt < 4; ++nt) W.b1v[nt] = b1[nt * 16 + nl];
    W.b2v = (nl < 4) ? b2[nl] : 0.f;
}

// Load a 16-bit A-fragment from an LDS row (ISA layout: halves 0..7 = K kb..kb+7,
// halves 8..15 = K kb+16..kb+23) via two 16-byte LDS loads.
__device__ __forceinline__ v16h ld_afrag(const _Float16* p) {
    const v8h lo = *(const v8h*)(p);
    const v8h hi = *(const v8h*)(p + 16);
    v16h a;
    #pragma unroll
    for (int h = 0; h < 8; ++h) { a[h] = lo[h]; a[h + 8] = hi[h]; }
    return a;
}

// One 16-sample tile through the MLP. Returns layer-2 accumulator (C layout:
// value at lane l, elem v is out[m = v + 8*(l>>4)][n = l&15], bias added).
__device__ v8f mlp_tile(const MLPW& W, _Float16* feat /*16x96*/, _Float16* hbuf /*16x64*/,
                        int lane, int s0, int S,
                        float ox, float oy, float oz, float dx, float dy, float dz) {
    // ---- build feature tile (f16) in this wave's LDS slice ----
    for (int i = lane; i < 16 * 96; i += 32) {
        const int m = i / 96, k = i - m * 96;
        const int s = s0 + m;
        const float t = NEAR_F + (FAR_F - NEAR_F) * (float)s / (float)(S - 1);
        const float p0 = ox + dx * t, p1 = oy + dy * t, p2 = oz + dz * t;
        float val;
        if (k < 3) {
            val = (k == 0) ? p0 : (k == 1 ? p1 : p2);
        } else if (k < 63) {
            const int j = k - 3, l = j / 6, c = j % 6;
            const float f = (float)(1 << l);
            const float x = (c % 3 == 0) ? p0 : (c % 3 == 1 ? p1 : p2);
            val = (c < 3) ? __sinf(f * x) : __cosf(f * x);
        } else if (k == 63) val = dx;
        else if (k == 64)   val = dy;
        else if (k == 65)   val = dz;
        else                val = 0.f;   // K padding 66..95
        feat[m * 96 + k] = (_Float16)val;
    }
    __syncthreads();

    const int m  = lane & 15;
    const int hi = lane >> 4;
    const int nl = lane & 15;

    v16h A[3];
    #pragma unroll
    for (int ks = 0; ks < 3; ++ks)
        A[ks] = ld_afrag(feat + m * 96 + ks * 32 + hi * 8);

    // ---- layer 1: 66->64, relu, write f16 hidden to LDS ----
    #pragma unroll
    for (int nt = 0; nt < 4; ++nt) {
        v8f acc;
        #pragma unroll
        for (int v = 0; v < 8; ++v) acc[v] = 0.f;
        #pragma unroll
        for (int ks = 0; ks < 3; ++ks)
            acc = __builtin_amdgcn_wmma_f32_16x16x32_f16(false, A[ks], false, W.B1[ks][nt],
                                                         (short)0, acc, false, false);
        #pragma unroll
        for (int v = 0; v < 8; ++v) {
            float hv = fmaxf(acc[v] + W.b1v[nt], 0.f);
            hbuf[(v + 8 * hi) * HID + nt * 16 + nl] = (_Float16)hv;
        }
    }
    __syncthreads();

    // ---- layer 2: 64->4 (N padded to 16) ----
    v16h A2[2];
    #pragma unroll
    for (int ks = 0; ks < 2; ++ks)
        A2[ks] = ld_afrag(hbuf + m * HID + ks * 32 + hi * 8);

    v8f acc2;
    #pragma unroll
    for (int v = 0; v < 8; ++v) acc2[v] = 0.f;
    acc2 = __builtin_amdgcn_wmma_f32_16x16x32_f16(false, A2[0], false, W.B2[0],
                                                  (short)0, acc2, false, false);
    acc2 = __builtin_amdgcn_wmma_f32_16x16x32_f16(false, A2[1], false, W.B2[1],
                                                  (short)0, acc2, false, false);
    #pragma unroll
    for (int v = 0; v < 8; ++v) acc2[v] += W.b2v;
    return acc2;
}

// ---------------------------------------------------------------------------
__global__ void nerf_init_kernel(unsigned int* maxbits) {
    if (threadIdx.x == 0 && blockIdx.x == 0) *maxbits = 0u;
}

// Coarse pass: MLP sigmas -> sigma2weights mask weights + global max
__global__ void __launch_bounds__(WPB * 32)
nerf_coarse_kernel(const float* __restrict__ rays_o, const float* __restrict__ rays_d,
                   const float* __restrict__ w1, const float* __restrict__ b1,
                   const float* __restrict__ w2, const float* __restrict__ b2,
                   const float* __restrict__ noise_mask,
                   float* __restrict__ wmask, unsigned int* __restrict__ maxbits) {
    __shared__ __align__(16) _Float16 s_feat[WPB][16 * 96];
    __shared__ __align__(16) _Float16 s_hbuf[WPB][16 * HID];
    __shared__ float s_sig[WPB][SC];

    const int lane = threadIdx.x & 31;
    const int wv   = threadIdx.x >> 5;
    const int ray  = blockIdx.x * WPB + wv;

    const float ox = rays_o[ray * 3 + 0], oy = rays_o[ray * 3 + 1], oz = rays_o[ray * 3 + 2];
    const float dx = rays_d[ray * 3 + 0], dy = rays_d[ray * 3 + 1], dz = rays_d[ray * 3 + 2];

    MLPW W;
    load_weights(W, w1, b1, w2, b2, lane);

    const int hi = lane >> 4, nl = lane & 15;
    for (int t = 0; t < SC / 16; ++t) {
        v8f o = mlp_tile(W, s_feat[wv], s_hbuf[wv], lane, t * 16, SC, ox, oy, oz, dx, dy, dz);
        if (nl == 0) {
            #pragma unroll
            for (int v = 0; v < 8; ++v) s_sig[wv][t * 16 + v + 8 * hi] = o[v];  // raw sigma
        }
    }

    if (lane == 0) {
        const float dzc = (FAR_F - NEAR_F) / (float)(SC - 1);
        float Tw = 1.f, lmax = 0.f;
        for (int s = 0; s < SC; ++s) {
            const float delta = (s == SC - 1) ? 1e10f : dzc;
            const float a  = 1.f - __expf(-delta * softplusf_(s_sig[wv][s] + noise_mask[ray * SC + s]));
            const float wt = a * Tw;
            Tw *= (1.f - a + 1e-10f);
            wmask[ray * SC + s] = wt;
            lmax = fmaxf(lmax, wt);
        }
        atomicMax(maxbits, __float_as_uint(lmax));  // weights >= 0: bit compare is monotone
    }
}

// Fine pass: masked MLP + full composite -> [r,g,b,depth,opacity]
__global__ void __launch_bounds__(WPB * 32)
nerf_fine_kernel(const float* __restrict__ rays_o, const float* __restrict__ rays_d,
                 const float* __restrict__ w1, const float* __restrict__ b1,
                 const float* __restrict__ w2, const float* __restrict__ b2,
                 const float* __restrict__ noise_f,
                 const float* __restrict__ wmask, const unsigned int* __restrict__ maxbits,
                 float* __restrict__ out) {
    __shared__ __align__(16) _Float16 s_feat[WPB][16 * 96];
    __shared__ __align__(16) _Float16 s_hbuf[WPB][16 * HID];
    __shared__ float s_sig[WPB][SF];
    __shared__ float s_rgb[WPB][SF * 3];

    const int lane = threadIdx.x & 31;
    const int wv   = threadIdx.x >> 5;
    const int ray  = blockIdx.x * WPB + wv;

    const float thresh = fminf(1e-4f, __uint_as_float(*maxbits));

    const float ox = rays_o[ray * 3 + 0], oy = rays_o[ray * 3 + 1], oz = rays_o[ray * 3 + 2];
    const float dx = rays_d[ray * 3 + 0], dy = rays_d[ray * 3 + 1], dz = rays_d[ray * 3 + 2];
    const float rlen = sqrtf(dx * dx + dy * dy + dz * dz);

    MLPW W;
    load_weights(W, w1, b1, w2, b2, lane);

    const int hi = lane >> 4, nl = lane & 15;
    for (int t = 0; t < SF / 16; ++t) {
        v8f o = mlp_tile(W, s_feat[wv], s_hbuf[wv], lane, t * 16, SF, ox, oy, oz, dx, dy, dz);
        if (nl < 4) {
            #pragma unroll
            for (int v = 0; v < 8; ++v) {
                const int s = t * 16 + v + 8 * hi;
                const bool msk = (wmask[ray * SC + (s >> 2)] >= thresh);
                if (nl == 0) s_sig[wv][s] = msk ? o[v] : -20.0f;                 // SIGMA_DEFAULT
                else         s_rgb[wv][s * 3 + (nl - 1)] = msk ? sigmoidf_(o[v]) : 1.0f;
            }
        }
    }

    if (lane == 0) {
        const float dzf = (FAR_F - NEAR_F) / (float)(SF - 1);
        float cum = 0.f, Tw = 1.f;
        float cr = 0.f, cg = 0.f, cb = 0.f, wsum = 0.f, depth = 0.f, opac = 0.f;
        for (int s = 0; s < SF; ++s) {
            const float zv    = NEAR_F + (FAR_F - NEAR_F) * (float)s / (float)(SF - 1);
            const float delta = (s == SF - 1) ? 1e10f : dzf;
            const float sg    = s_sig[wv][s];
            // depth/opacity chain: dist = delta*|d|, no noise
            const float sd    = softplusf_(sg) * (delta * rlen);
            const float alpha = 1.f - __expf(-sd);
            const float T     = __expf(-cum);
            const float prob  = T * alpha;
            opac  += prob;
            depth += zv * prob;
            cum   += sd;
            // rgb chain: sigma2weights with deltas (no |d|) and noise
            const float aw = 1.f - __expf(-delta * softplusf_(sg + noise_f[ray * SF + s]));
            const float wt = aw * Tw;
            Tw *= (1.f - aw + 1e-10f);
            wsum += wt;
            cr += wt * s_rgb[wv][s * 3 + 0];
            cg += wt * s_rgb[wv][s * 3 + 1];
            cb += wt * s_rgb[wv][s * 3 + 2];
        }
        const float wb = 1.f - wsum;  // WHITE_BACK
        out[ray * 5 + 0] = cr + wb;
        out[ray * 5 + 1] = cg + wb;
        out[ray * 5 + 2] = cb + wb;
        out[ray * 5 + 3] = depth;
        out[ray * 5 + 4] = opac;
    }
}

extern "C" void kernel_launch(void* const* d_in, const int* in_sizes, int n_in,
                              void* d_out, int out_size, void* d_ws, size_t ws_size,
                              hipStream_t stream) {
    const float* rays_o = (const float*)d_in[0];
    const float* rays_d = (const float*)d_in[1];
    const float* w1_c   = (const float*)d_in[2];
    const float* b1_c   = (const float*)d_in[3];
    const float* w2_c   = (const float*)d_in[4];
    const float* b2_c   = (const float*)d_in[5];
    const float* w1_f   = (const float*)d_in[6];
    const float* b1_f   = (const float*)d_in[7];
    const float* w2_f   = (const float*)d_in[8];
    const float* b2_f   = (const float*)d_in[9];
    // d_in[10] = noise_c_inf: dead in the reference (coarse composite outputs unused)
    const float* noise_c_mask = (const float*)d_in[11];
    const float* noise_f_inf  = (const float*)d_in[12];

    unsigned int* maxbits = (unsigned int*)d_ws;
    float* wmask = (float*)((char*)d_ws + 64);   // 4096*64 floats

    const dim3 blk(WPB * 32);
    const dim3 grd(NRAYS / WPB);

    nerf_init_kernel<<<1, 1, 0, stream>>>(maxbits);
    nerf_coarse_kernel<<<grd, blk, 0, stream>>>(rays_o, rays_d, w1_c, b1_c, w2_c, b2_c,
                                                noise_c_mask, wmask, maxbits);
    nerf_fine_kernel<<<grd, blk, 0, stream>>>(rays_o, rays_d, w1_f, b1_f, w2_f, b2_f,
                                              noise_f_inf, wmask, maxbits, (float*)d_out);
}
